// Point2Image_43516608643709
// MI455X (gfx1250) — matrix operand order repaired and checked by reading, not compiled, not run on hardware
//
#include <hip/hip_runtime.h>
#include <hip/hip_bf16.h>

typedef __attribute__((ext_vector_type(16))) _Float16 v16h;
typedef __attribute__((ext_vector_type(8)))  _Float16 v8h;
typedef __attribute__((ext_vector_type(8)))  float    v8f;

#define RES   384
#define NPTS  1024
#define KDIM  1024          // K dimension of the GEMMs = number of points
#define HW_K  6             // round(3*0.005*384)
#define HW_F  23            // round(3*0.02*384)
#define SLOT  (RES * KDIM)  // one factor matrix: 384 x 1024 f16

// ---------------------------------------------------------------------------
// Phase 1: build six transposed factor matrices in f16, WMMA-ready layout.
//   slot 0: GykT[r][n] = exp(-(lin_r - y_n)^2 * 20000) * mask_k
//   slot 1: GxkT[c][n]
//   slot 2: GxfT[c][n]
//   slot 3..5: AFf[r][n] = feat_f[n] * GyfT[r][n]   (feature folded into A)
// Row-major [res][N] so every WMMA fragment load is contiguous.
// ---------------------------------------------------------------------------
__global__ __launch_bounds__(256)
void p2i_gauss(const float* __restrict__ p, _Float16* __restrict__ ws) {
  const int n = blockIdx.x * blockDim.x + threadIdx.x;   // point id 0..1023
  const int r = blockIdx.y;                              // grid cell 0..383

  const float x  = p[n * 5 + 0];
  const float y  = p[n * 5 + 1];
  const float f0 = p[n * 5 + 2];
  const float f1 = p[n * 5 + 3];
  const float f2 = p[n * 5 + 4];

  const float lin = (float)r * (1.0f / 383.0f);          // linspace(0,1,384)
  const int   cx  = (int)floorf(x * (float)RES);
  const int   cy  = (int)floorf(y * (float)RES);
  const float dx  = lin - x;
  const float dy  = lin - y;

  const float inv2s2_k = 20000.0f;   // 1/(2*0.005^2)
  const float inv2s2_f = 1250.0f;    // 1/(2*0.02^2)

  float gyk = (r >= cy - HW_K && r <= cy + HW_K) ? expf(-dy * dy * inv2s2_k) : 0.0f;
  float gxk = (r >= cx - HW_K && r <= cx + HW_K) ? expf(-dx * dx * inv2s2_k) : 0.0f;
  float gyf = (r >= cy - HW_F && r <= cy + HW_F) ? expf(-dy * dy * inv2s2_f) : 0.0f;
  float gxf = (r >= cx - HW_F && r <= cx + HW_F) ? expf(-dx * dx * inv2s2_f) : 0.0f;

  const int o = r * KDIM + n;                            // coalesced over n
  ws[0 * SLOT + o] = (_Float16)gyk;
  ws[1 * SLOT + o] = (_Float16)gxk;
  ws[2 * SLOT + o] = (_Float16)gxf;
  ws[3 * SLOT + o] = (_Float16)(gyf * f0);
  ws[4 * SLOT + o] = (_Float16)(gyf * f1);
  ws[5 * SLOT + o] = (_Float16)(gyf * f2);
}

// ---------------------------------------------------------------------------
// Phase 2: 4 GEMMs, each 384x384 = (384xK) * (Kx384), K=1024.
// One wave32 per 32x32 output block: 2 A-frags x 2 B-frags -> 4 f32
// accumulators, 4 x v_wmma_f32_16x16x32_f16 per K-step (32 steps).
// Register blocking halves L2 traffic vs 16x16-per-wave and gives a
// 4-WMMA : 8-load inner loop for better latency hiding.
// ---------------------------------------------------------------------------
union ABfrag { v16h v; v8h h[2]; };

__device__ __forceinline__ v8f wmma_step(const v16h& a, const v16h& b, const v8f& c) {
  return __builtin_amdgcn_wmma_f32_16x16x32_f16(
      /*neg_a=*/false, a, /*neg_b=*/false, b,
      /*c_mod=*/(short)0, c, /*reuse_a=*/false, /*reuse_b=*/false);
}

__global__ __launch_bounds__(32)
void p2i_wmma(const _Float16* __restrict__ ws, float* __restrict__ out) {
  const int tile = blockIdx.x;            // 0 .. 4*12*12-1
  const int ch   = tile / 144;            // output channel 0..3
  const int t    = tile % 144;
  const int tr   = t / 12;                // 32-row block
  const int tc   = t % 12;                // 32-col block

  // channel 0: density = GykT x Gxk ; channels 1..3: AFf x Gxf
  const _Float16* A = ws + (size_t)((ch == 0) ? 0 : (2 + ch)) * SLOT;
  const _Float16* B = ws + (size_t)((ch == 0) ? 1 : 2) * SLOT;

  const int lane = threadIdx.x & 31;
  const int hi   = lane >> 4;             // lane half: 0 or 1
  const int l    = lane & 15;

  // A fragment rows: lane half 0 -> K 0..15; half 1 -> K 16..31
  const _Float16* Ap0 = A + (size_t)(tr * 32 + l) * KDIM + hi * 16;
  const _Float16* Ap1 = Ap0 + (size_t)16 * KDIM;
  // B fragment (transposed-B rows): half 0 -> K 0..15; half 1 -> K 16..31
  const _Float16* Bp0 = B + (size_t)(tc * 32 + l) * KDIM + hi * 16;
  const _Float16* Bp1 = Bp0 + (size_t)16 * KDIM;

  v8f acc00 = {}, acc01 = {}, acc10 = {}, acc11 = {};
#pragma unroll 2
  for (int k = 0; k < KDIM; k += 32) {
    ABfrag a0, a1;
    a0.h[0] = *(const v8h*)(Ap0 + k);
    a0.h[1] = *(const v8h*)(Ap0 + k + 8);
    a1.h[0] = *(const v8h*)(Ap1 + k);
    a1.h[1] = *(const v8h*)(Ap1 + k + 8);
    v16h b0 = *(const v16h*)(Bp0 + k);
    v16h b1 = *(const v16h*)(Bp1 + k);
    acc00 = wmma_step(a0.v, b0, acc00);
    acc01 = wmma_step(a0.v, b1, acc01);
    acc10 = wmma_step(a1.v, b0, acc10);
    acc11 = wmma_step(a1.v, b1, acc11);
  }

  // C/D layout: VGPR i holds row (i + 8*hi), column = lane%16
  float* O0 = out + (size_t)ch * (RES * RES)
                  + (size_t)(tr * 32 + hi * 8) * RES + tc * 32 + l;
  float* O1 = O0 + (size_t)16 * RES;
#pragma unroll
  for (int i = 0; i < 8; ++i) {
    O0[i * RES]      = acc00[i];
    O0[i * RES + 16] = acc01[i];
    O1[i * RES]      = acc10[i];
    O1[i * RES + 16] = acc11[i];
  }
}

// ---------------------------------------------------------------------------
extern "C" void kernel_launch(void* const* d_in, const int* in_sizes, int n_in,
                              void* d_out, int out_size, void* d_ws, size_t ws_size,
                              hipStream_t stream) {
  const float* p   = (const float*)d_in[0];    // [1024, 5] f32
  float*       out = (float*)d_out;            // [1, 4, 384, 384] f32
  _Float16*    ws  = (_Float16*)d_ws;          // 6 * 384*1024 f16 = 4.5 MB

  dim3 b1(256), g1(NPTS / 256, RES);
  p2i_gauss<<<g1, b1, 0, stream>>>(p, ws);

  p2i_wmma<<<4 * 12 * 12, 32, 0, stream>>>(ws, out);
}
